// GCGM_9345848836779
// MI455X (gfx1250) — matrix operand order repaired and checked by reading, not compiled, not run on hardware
//
#include <hip/hip_runtime.h>
#include <stdint.h>

// ---------------- problem constants ----------------
#define B_DIM 8
#define C_DIM 256
#define H_DIM 128
#define W_DIM 128
#define HW    (H_DIM * W_DIM)     // 16384

// ---------------- tiling ----------------
#define MT  64    // output channels per channel-tile
#define NCT 4     // channel tiles per workgroup (covers all 256 out channels)
#define NT  256   // pixels per WG (2 full rows of W => 64 complete 2x2 patches)
#define KC  32    // K per wmma step
#define NKC (C_DIM / KC)          // 8 chunks

typedef __attribute__((ext_vector_type(16))) __bf16        v16bf;
typedef __attribute__((ext_vector_type(8)))  float         v8f;
typedef __attribute__((ext_vector_type(4)))  float         f32x4;
typedef __attribute__((ext_vector_type(4)))  unsigned int  u32x4;

union Frag16 { v16bf bf; u32x4 q[2]; };

// two fp32 -> packed 2x bf16 in one u32 (lo = a, hi = b), single native instruction
// (gfx1250 has v_cvt_pk_bf16_f32 -- confirmed in previous round's codegen)
__device__ __forceinline__ unsigned int cvt2bf(float a, float b) {
    unsigned int u;
    asm("v_cvt_pk_bf16_f32 %0, %1, %2" : "=v"(u) : "v"(a), "v"(b));
    return u;
}

__global__ __launch_bounds__(256)
void fused_gcgm_kernel(const float* __restrict__ x1, const float* __restrict__ x2,
                       const float* __restrict__ x3,
                       const float* __restrict__ wx1, const float* __restrict__ bx1,
                       const float* __restrict__ wx2, const float* __restrict__ bx2,
                       const float* __restrict__ wy,  const float* __restrict__ by,
                       const float* __restrict__ t0w1, const float* __restrict__ t0w2,
                       const float* __restrict__ t1w1, const float* __restrict__ t1w2,
                       const float* __restrict__ t2w1, const float* __restrict__ t2w2,
                       float* __restrict__ out)
{
    // ~289 KiB LDS of the 320 KiB/WGP -> 1 WG/WGP; x-slab converted to bf16 ONCE
    __shared__ unsigned short Bfull[NT][C_DIM];       // resident x slab, n-major (128K)
    __shared__ unsigned short Alds[MT][C_DIM];        // weight tile bf16 (32K)
    __shared__ float          accLds[MT][NT];         // GEMM staging (64K)
    __shared__ float          sFull[NCT * MT][NT/4];  // gate per (channel, patch) (64K)
    __shared__ float          matLds[3][16];
    __shared__ float          biasLds[MT];

    const int tid  = (int)threadIdx.x;
    const int lane = tid & 31;
    const int wv   = tid >> 5;
    const int l16  = lane & 15;
    const int half = lane >> 4;

    const int hp = (int)blockIdx.x;     // row-pair 0..63
    const int b  = (int)blockIdx.y;     // batch 0..7

    const long xoff = (long)b * C_DIM * HW + (long)hp * NT;  // + c*HW + n

    if (tid < 48) {
        const float* w1 = (tid < 16) ? t0w1 : (tid < 32) ? t1w1 : t2w1;
        const float* w2 = (tid < 16) ? t0w2 : (tid < 32) ? t1w2 : t2w2;
        const int e = tid & 15, o = e >> 2, i = e & 3;
        float m = 0.f;
        #pragma unroll
        for (int h = 0; h < 8; ++h) m += w2[o * 8 + h] * w1[h * 4 + i];
        matLds[tid >> 4][e] = m;
    }

    const float* Xs[3] = { x1, x2, x3 };
    const float* Ws[3] = { wx1, wx2, wy };
    const float* Bs[3] = { bx1, bx2, by };

    // wave -> output-tile mapping: mt = wv>>1 (16 rows), 8 N-tiles each
    const int mt  = wv >> 1;
    const int nt0 = (wv & 1) * 8;
    // slab staging: 2 adjacent K rows per thread, n fastest across lanes (coalesced)
    const int skk = (tid >> 4) * 2;
    const int sn  = (tid & 15) * 16;

    #pragma unroll
    for (int phase = 0; phase < 3; ++phase) {
        const float* X = Xs[phase];

        // stage one 32x256 K-chunk of the slab (fp32 -> bf16, transposed, pair-packed)
        auto slabChunk = [&](int k0) {
            const float* s0 = X + xoff + (long)(k0 + skk) * HW + sn;
            const float* s1 = s0 + HW;
            if (k0 + KC < C_DIM) __builtin_prefetch(s0 + (long)KC * HW, 0, 1);
            #pragma unroll
            for (int j = 0; j < 4; ++j) {
                f32x4 a = *(const f32x4*)(s0 + j * 4);
                f32x4 c = *(const f32x4*)(s1 + j * 4);
                *(unsigned int*)&Bfull[sn + j * 4 + 0][k0 + skk] = cvt2bf(a.x, c.x);
                *(unsigned int*)&Bfull[sn + j * 4 + 1][k0 + skk] = cvt2bf(a.y, c.y);
                *(unsigned int*)&Bfull[sn + j * 4 + 2][k0 + skk] = cvt2bf(a.z, c.z);
                *(unsigned int*)&Bfull[sn + j * 4 + 3][k0 + skk] = cvt2bf(a.w, c.w);
            }
        };
        // stage weight tile [ct*64, ct*64+64) x 256 + bias slice
        auto stageA = [&](int ct) {
            const int row = tid >> 2;
            const int c0  = (tid & 3) * 64;
            const float* src = Ws[phase] + (long)(ct * MT + row) * C_DIM + c0;
            u32x4* dst = (u32x4*)&Alds[row][c0];
            #pragma unroll
            for (int j = 0; j < 8; ++j) {
                f32x4 v0 = *(const f32x4*)(src + j * 8);
                f32x4 v1 = *(const f32x4*)(src + j * 8 + 4);
                u32x4 o;
                o.x = cvt2bf(v0.x, v0.y); o.y = cvt2bf(v0.z, v0.w);
                o.z = cvt2bf(v1.x, v1.y); o.w = cvt2bf(v1.z, v1.w);
                dst[j] = o;
            }
            if (tid < MT) biasLds[tid] = Bs[phase][ct * MT + tid];
        };
        // one K-chunk of WMMAs against the resident slab
        auto gemmChunk = [&](int k0, v8f* acc) {
            Frag16 afrag;  // A 16x32 bf16: lo lanes K{0..7,16..23}, hi K{8..15,24..31}
            const int arow = mt * 16 + l16;
            const int akb  = half * 8;
            afrag.q[0] = *(const u32x4*)&Alds[arow][k0 + akb];
            afrag.q[1] = *(const u32x4*)&Alds[arow][k0 + akb + 16];
            #pragma unroll
            for (int j = 0; j < 8; ++j) {
                Frag16 bfrag;  // B 32x16 bf16: lo lanes K{0..7,16..23}, hi K{8..15,24..31}, col l16
                const int ncol = (nt0 + j) * 16 + l16;
                const int bkb  = half * 8;
                bfrag.q[0] = *(const u32x4*)&Bfull[ncol][k0 + bkb];
                bfrag.q[1] = *(const u32x4*)&Bfull[ncol][k0 + bkb + 16];
                acc[j] = __builtin_amdgcn_wmma_f32_16x16x32_bf16(
                            false, afrag.bf, false, bfrag.bf,
                            (short)0, acc[j], false, false);
            }
        };
        // acc -> LDS, then 4x4 patch MLP + gating for one channel tile
        auto finishTile = [&](int ct, v8f* acc) {
            #pragma unroll
            for (int j = 0; j < 8; ++j) {
                const int ncol = (nt0 + j) * 16 + l16;
                #pragma unroll
                for (int r = 0; r < 8; ++r)
                    accLds[mt * 16 + half * 8 + r][ncol] = acc[j][r];
            }
            __syncthreads();
            float M[16];
            #pragma unroll
            for (int e = 0; e < 16; ++e) M[e] = matLds[phase][e];
            for (int j = 0; j < 16; ++j) {
                const int task = j * 256 + tid;
                const int m = task >> 6;
                const int p = task & 63;
                const float bias = biasLds[m];
                const float2 a01 = *(const float2*)&accLds[m][2 * p];
                const float2 a23 = *(const float2*)&accLds[m][128 + 2 * p];
                const float v0 = a01.x + bias, v1 = a01.y + bias;
                const float v2 = a23.x + bias, v3 = a23.y + bias;
                const float r0 = fmaxf(M[0]  * v0 + M[1]  * v1 + M[2]  * v2 + M[3]  * v3, 0.f);
                const float r1 = fmaxf(M[4]  * v0 + M[5]  * v1 + M[6]  * v2 + M[7]  * v3, 0.f);
                const float r2 = fmaxf(M[8]  * v0 + M[9]  * v1 + M[10] * v2 + M[11] * v3, 0.f);
                const float r3 = fmaxf(M[12] * v0 + M[13] * v1 + M[14] * v2 + M[15] * v3, 0.f);
                float* sp = &sFull[ct * MT + m][p];
                if (phase == 0) {
                    *sp = r0 + r1 + r2 + r3;
                } else if (phase == 1) {
                    *sp *= (r0 + r1 + r2 + r3);
                } else {
                    const float s = *sp;
                    float* o = out + xoff + (long)(ct * MT + m) * HW;
                    float2 lo; lo.x = r0 * s; lo.y = r1 * s;
                    float2 hi; hi.x = r2 * s; hi.y = r3 * s;
                    *(float2*)(o + 2 * p)       = lo;
                    *(float2*)(o + 128 + 2 * p) = hi;
                }
            }
            __syncthreads();   // before Alds/accLds reuse
        };

        v8f acc[8];

        // ---- channel tile 0: GEMM pipelined against slab staging ----
        #pragma unroll
        for (int t = 0; t < 8; ++t)
            #pragma unroll
            for (int r = 0; r < 8; ++r) acc[t][r] = 0.f;
        stageA(0);
        slabChunk(0);
        __syncthreads();
        for (int i = 0; i < NKC; ++i) {
            if (i < NKC - 1) slabChunk((i + 1) * KC);
            gemmChunk(i * KC, acc);
            __syncthreads();   // chunk i+1 fully staged before it is consumed
        }
        finishTile(0, acc);

        // ---- channel tiles 1..3: barrier-free GEMM on the resident slab ----
        for (int ct = 1; ct < NCT; ++ct) {
            stageA(ct);
            #pragma unroll
            for (int t = 0; t < 8; ++t)
                #pragma unroll
                for (int r = 0; r < 8; ++r) acc[t][r] = 0.f;
            __syncthreads();   // Alds staged
            #pragma unroll 2
            for (int i = 0; i < NKC; ++i) gemmChunk(i * KC, acc);
            finishTile(ct, acc);
        }
    }
}

extern "C" void kernel_launch(void* const* d_in, const int* in_sizes, int n_in,
                              void* d_out, int out_size, void* d_ws, size_t ws_size,
                              hipStream_t stream) {
    (void)in_sizes; (void)n_in; (void)d_ws; (void)ws_size; (void)out_size;
    const float* x1   = (const float*)d_in[0];
    const float* x2   = (const float*)d_in[1];
    const float* x3   = (const float*)d_in[2];
    const float* wx1  = (const float*)d_in[3];
    const float* bx1  = (const float*)d_in[4];
    const float* wx2  = (const float*)d_in[5];
    const float* bx2  = (const float*)d_in[6];
    const float* wy   = (const float*)d_in[7];
    const float* by   = (const float*)d_in[8];
    const float* t0w1 = (const float*)d_in[9];
    const float* t0w2 = (const float*)d_in[10];
    const float* t1w1 = (const float*)d_in[11];
    const float* t1w2 = (const float*)d_in[12];
    const float* t2w1 = (const float*)d_in[13];
    const float* t2w2 = (const float*)d_in[14];
    float* o = (float*)d_out;

    dim3 grid(H_DIM / 2, B_DIM);   // (64 row-pairs, 8 batches) = 512 WGs
    dim3 block(256);
    fused_gcgm_kernel<<<grid, block, 0, stream>>>(x1, x2, x3, wx1, bx1, wx2, bx2,
                                                  wy, by, t0w1, t0w2, t1w1, t1w2,
                                                  t2w1, t2w2, o);
}
